// DeepseekV3DecoderLayer_33638183862742
// MI455X (gfx1250) — compile-verified, matrix operand order checked by reading, not run on hardware
//
#include <hip/hip_runtime.h>

// ---------------------------------------------------------------------------
// DeepSeek-V3 MLA decoder layer for gfx1250 (MI455X).
// bf16 WMMA GEMMs, async global->LDS staging (CDNA5 GLOBAL_LOAD_ASYNC_TO_LDS),
// double-buffered LDS, flash-style MLA attention.
// ---------------------------------------------------------------------------

#define DEV static __device__ __forceinline__

typedef __attribute__((ext_vector_type(16))) __bf16 v16bf;
typedef __attribute__((ext_vector_type(8)))  __bf16 v8bf;
typedef __attribute__((ext_vector_type(8)))  float  v8f;
typedef __attribute__((ext_vector_type(8)))  unsigned short v8us;

DEV unsigned short f2bf(float f) {
  unsigned int u = __float_as_uint(f);
  unsigned int r = u + 0x7FFFu + ((u >> 16) & 1u);   // round-to-nearest-even
  return (unsigned short)(r >> 16);
}
DEV float bf2f(unsigned short h) { return __uint_as_float(((unsigned int)h) << 16); }

union FragU { v16bf v; v8bf h[2]; };

// CDNA5 async copy: memory -> LDS, 16B per lane, tracked with ASYNCcnt.
// dst_lds = low 32 bits of the generic LDS address (aperture bits discarded).
DEV void async_ld_b128(unsigned dst_lds, unsigned long long src) {
  asm volatile("global_load_async_to_lds_b128 %0, %1, off"
               :: "v"(dst_lds), "v"(src) : "memory");
}
DEV void async_ld_b128_o16(unsigned dst_lds, unsigned long long src) {
  asm volatile("global_load_async_to_lds_b128 %0, %1, off offset:16"
               :: "v"(dst_lds), "v"(src) : "memory");
}
DEV void wait_async0() { asm volatile("s_wait_asynccnt 0x0" ::: "memory"); }
DEV void wait_async4() { asm volatile("s_wait_asynccnt 0x4" ::: "memory"); }

DEV unsigned lds_off(const void* p) { return (unsigned)(unsigned long long)p; }

// Fragment layout per CDNA5 ISA "16-bit A-Matrix 16x32":
// lane L: row = (L&15), khalf = L>>4; chunk0 @ K=khalf*8, chunk1 @ K=16+khalf*8.
// B operand uses the same layout when B is stored N-major (i.e. B^T row-major).
DEV v16bf load_frag(const unsigned short* base, int ld, int row, int k0) {
  const int lane = threadIdx.x & 31;
  const unsigned short* p =
      base + (size_t)(row + (lane & 15)) * (size_t)ld + k0 + ((lane >> 4) << 3);
  FragU f;
  f.h[0] = *(const v8bf*)(p);
  f.h[1] = *(const v8bf*)(p + 16);
  return f.v;
}

DEV v8f wmma_bf16(v16bf a, v16bf b, v8f c) {
  // (neg_a, A, neg_b, B, c_mod, C, reuse_a, reuse_b)
  return __builtin_amdgcn_wmma_f32_16x16x32_bf16(false, a, false, b, (short)0, c,
                                                 false, false);
}

DEV v8f vzero8() {
  v8f z;
#pragma unroll
  for (int i = 0; i < 8; ++i) z[i] = 0.0f;
  return z;
}

// ---------------------------------------------------------------------------
// Generic GEMM: C[M,N] = A[M,K] (bf16, lda) * Bt[N,K] (bf16, ldb; = B^T).
// Block tile 128x128, 8 waves (2 x 4), wave tile 64x32, K-step 32.
// Double-buffered LDS; tiles staged with GLOBAL_LOAD_ASYNC_TO_LDS_B128.
// M must be a multiple of 128; K a multiple of 32; N arbitrary (guarded).
// ---------------------------------------------------------------------------
#define GT_K 32
#define LDS_LD 40   // 32 + 8 halves pad; row stride 80B (16B aligned)

__global__ __launch_bounds__(256) void gemm_bf16_kernel(
    const unsigned short* __restrict__ A, int lda,
    const unsigned short* __restrict__ Bt, int ldb,
    void* __restrict__ Cv, int ldc,
    int M, int N, int K, int storeBf16)
{
  __shared__ unsigned short As[2][128 * LDS_LD];
  __shared__ unsigned short Bs[2][128 * LDS_LD];
  __shared__ unsigned short Trash[32];          // sink for OOB async lanes

  const int tid  = threadIdx.x;
  const int lane = tid & 31;
  const int wave = tid >> 5;
  const int wm = wave >> 2;           // 0..1
  const int wn = wave & 3;            // 0..3
  const int bm = blockIdx.y * 128;
  const int bn = blockIdx.x * 128;

  v8f acc[4][2];
#pragma unroll
  for (int i = 0; i < 4; ++i)
#pragma unroll
    for (int j = 0; j < 2; ++j) acc[i][j] = vzero8();

  const int lr = tid >> 1;            // 0..127: tile row loaded by this thread
  const int lc = (tid & 1) << 4;      // 0 or 16 (halves)

  const unsigned short* pa = A + (size_t)(bm + lr) * (size_t)lda + lc;
  const int nrow = bn + lr;
  const bool bok = (nrow < N);
  // OOB lanes read a valid row and dump into the trash slot so every wave
  // issues exactly 4 async ops per stage (keeps the wait threshold uniform).
  const unsigned short* pb = Bt + (size_t)(bok ? nrow : 0) * (size_t)ldb + lc;

  const unsigned aoff0 = lds_off(&As[0][lr * LDS_LD + lc]);
  const unsigned aoff1 = lds_off(&As[1][lr * LDS_LD + lc]);
  const unsigned boffr0 = lds_off(&Bs[0][lr * LDS_LD + lc]);
  const unsigned boffr1 = lds_off(&Bs[1][lr * LDS_LD + lc]);
  const unsigned toff = lds_off(&Trash[0]);
  const unsigned boff0 = bok ? boffr0 : toff;
  const unsigned boff1 = bok ? boffr1 : toff;

  // Pre-zero OOB B rows once (never overwritten by the redirected asyncs).
  if (!bok) {
    v8us z;
#pragma unroll
    for (int e = 0; e < 8; ++e) z[e] = 0;
    *(v8us*)&Bs[0][lr * LDS_LD + lc]     = z;
    *(v8us*)&Bs[0][lr * LDS_LD + lc + 8] = z;
    *(v8us*)&Bs[1][lr * LDS_LD + lc]     = z;
    *(v8us*)&Bs[1][lr * LDS_LD + lc + 8] = z;
  }

  auto stage = [&](int k0, int buf) {
    const unsigned long long ga = (unsigned long long)(pa + k0);
    const unsigned long long gb = (unsigned long long)(pb + k0);
    const unsigned ao = buf ? aoff1 : aoff0;
    const unsigned bo = buf ? boff1 : boff0;
    async_ld_b128(ao, ga);
    async_ld_b128_o16(ao, ga);
    async_ld_b128(bo, gb);
    async_ld_b128_o16(bo, gb);
  };

  stage(0, 0);
  const int nk = K / GT_K;

  for (int kk = 0; kk < nk; ++kk) {
    const int buf = kk & 1;
    if (kk + 1 < nk) {
      stage((kk + 1) * GT_K, buf ^ 1);
      wait_async4();                  // only the 4 just-issued may remain
    } else {
      wait_async0();
    }
    __syncthreads();

    const unsigned short* as = &As[buf][0];
    const unsigned short* bs = &Bs[buf][0];
    v16bf bf0 = load_frag(bs, LDS_LD, wn * 32 + 0, 0);
    v16bf bf1 = load_frag(bs, LDS_LD, wn * 32 + 16, 0);
#pragma unroll
    for (int i = 0; i < 4; ++i) {
      v16bf af = load_frag(as, LDS_LD, wm * 64 + i * 16, 0);
      acc[i][0] = wmma_bf16(af, bf0, acc[i][0]);
      acc[i][1] = wmma_bf16(af, bf1, acc[i][1]);
    }
    __syncthreads();                  // protect buf before it is re-staged
  }

  // C/D layout: lane holds col n=(lane&15); rows m = v + 8*(lane>>4)
  const int rb = 8 * (lane >> 4);
#pragma unroll
  for (int i = 0; i < 4; ++i) {
#pragma unroll
    for (int j = 0; j < 2; ++j) {
      const int col = bn + wn * 32 + j * 16 + (lane & 15);
      if (col < N) {
        const int rowb = bm + wm * 64 + i * 16 + rb;
#pragma unroll
        for (int v = 0; v < 8; ++v) {
          const size_t o = (size_t)(rowb + v) * (size_t)ldc + col;
          if (storeBf16) ((unsigned short*)Cv)[o] = f2bf(acc[i][j][v]);
          else           ((float*)Cv)[o]          = acc[i][j][v];
        }
      }
    }
  }
}

// ---------------------------------------------------------------------------
// Flash-style MLA attention.
//   Qf:  (T, 16, 576) bf16   q_full
//   Kf:  (T, 576)     bf16   k_full  (rows are N-major for B operand)
//   Ct:  (512, T)     bf16   compressed^T
//   Ol:  (T, 16, 512) bf16   o_latent out (already / l_i)
// Block: one head x 32 query rows; 8 waves. K processed in 64-wide blocks.
// ---------------------------------------------------------------------------
#define AQ 32
#define AK 64
#define QS_LD 584   // 576 + 8
#define SS_LD 68
#define PS_LD 72

__global__ __launch_bounds__(256) void mla_attn_kernel(
    const unsigned short* __restrict__ Qf,
    const unsigned short* __restrict__ Kf,
    const unsigned short* __restrict__ Ct,
    unsigned short* __restrict__ Ol,
    int T, float scale)
{
  const int h  = blockIdx.y;
  const int q0 = blockIdx.x * AQ;
  const int tid  = threadIdx.x;
  const int lane = tid & 31;
  const int wave = tid >> 5;

  __shared__ unsigned short Qs[AQ * QS_LD];
  __shared__ float          Ss[AQ * SS_LD];
  __shared__ unsigned short Ps[AQ * PS_LD];
  __shared__ float rowmax[AQ], rowsum[AQ], alpha[AQ];

  // stage Q tile (32 x 576 bf16) into LDS via async copies, 16B chunks
  for (int idx = tid; idx < AQ * 72; idx += 256) {
    const int r = idx / 72;
    const int c = (idx - r * 72) * 8;
    async_ld_b128(lds_off(&Qs[r * QS_LD + c]),
                  (unsigned long long)&Qf[((size_t)(q0 + r) * 16 + h) * 576 + c]);
  }
  if (tid < AQ) { rowmax[tid] = -3.0e38f; rowsum[tid] = 0.0f; }
  wait_async0();

  const int om = wave >> 2;   // O rows:  om*16 .. +16
  const int on = wave & 3;    // O cols:  on*128 .. +128
  const int sm = wave >> 2;   // S tile row
  const int sn = wave & 3;    // S tile col

  v8f o[8];
#pragma unroll
  for (int j = 0; j < 8; ++j) o[j] = vzero8();

  const int nkb = (q0 + AQ - 1) / AK + 1;   // causal coverage

  for (int kb = 0; kb < nkb; ++kb) {
    const int kbase = kb * AK;
    __syncthreads();

    // ---- S tile: 16x16 per wave, K = 576 ----
    v8f s = vzero8();
#pragma unroll 3
    for (int kd = 0; kd < 576; kd += 32) {
      v16bf aq = load_frag(Qs, QS_LD, sm * 16, kd);
      v16bf bk = load_frag(Kf, 576, kbase + sn * 16, kd);
      s = wmma_bf16(aq, bk, s);
    }
    {
      const int col = sn * 16 + (lane & 15);
      const int rbs = sm * 16 + 8 * (lane >> 4);
#pragma unroll
      for (int v = 0; v < 8; ++v) Ss[(rbs + v) * SS_LD + col] = s[v] * scale;
    }
    __syncthreads();

    // ---- online softmax, one thread per query row ----
    if (tid < AQ) {
      const int qg = q0 + tid;
      const float mo = rowmax[tid];
      float mx = mo;
      for (int c = 0; c < AK; ++c) {
        const int kg = kbase + c;
        if (kg <= qg) mx = fmaxf(mx, Ss[tid * SS_LD + c]);
      }
      const float al = __expf(mo - mx);
      float sum = 0.0f;
      for (int c = 0; c < AK; ++c) {
        const int kg = kbase + c;
        float p = 0.0f;
        if (kg <= qg) p = __expf(Ss[tid * SS_LD + c] - mx);
        Ps[tid * PS_LD + c] = f2bf(p);
        sum += p;
      }
      rowsum[tid] = rowsum[tid] * al + sum;
      rowmax[tid] = mx;
      alpha[tid]  = al;
    }
    __syncthreads();

    // ---- rescale running O by alpha ----
    {
      const int rbo = om * 16 + 8 * (lane >> 4);
#pragma unroll
      for (int v = 0; v < 8; ++v) {
        const float a = alpha[rbo + v];
#pragma unroll
        for (int j = 0; j < 8; ++j) o[j][v] *= a;
      }
    }

    // ---- O += P (16x64) x Ct^T (64 x 128 slice) ----
#pragma unroll
    for (int j = 0; j < 8; ++j) {
      const int rcol = on * 128 + j * 16;
#pragma unroll
      for (int ks = 0; ks < AK; ks += 32) {
        v16bf ap = load_frag(Ps, PS_LD, om * 16, ks);
        v16bf bc = load_frag(Ct, T, rcol, kbase + ks);
        o[j] = wmma_bf16(ap, bc, o[j]);
      }
    }
  }
  __syncthreads();

  // ---- finalize: divide by l_i, emit bf16 o_latent ----
  const int rbf = om * 16 + 8 * (lane >> 4);
#pragma unroll
  for (int v = 0; v < 8; ++v) {
    const float inv = 1.0f / rowsum[rbf + v];
    const size_t rowoff = ((size_t)(q0 + rbf + v) * 16 + h) * 512;
#pragma unroll
    for (int j = 0; j < 8; ++j) {
      const int col = on * 128 + j * 16 + (lane & 15);
      Ol[rowoff + col] = f2bf(o[j][v] * inv);
    }
  }
}

// ---------------------------------------------------------------------------
// Elementwise / reduction kernels
// ---------------------------------------------------------------------------
__global__ __launch_bounds__(256) void rmsnorm_kernel(
    const float* __restrict__ src, int sld,
    const float* __restrict__ w,
    unsigned short* __restrict__ dstB, int bld, int D)
{
  __shared__ float red[256];
  const int row = blockIdx.x;
  const float* x = src + (size_t)row * sld;
  float ss = 0.0f;
  for (int i = threadIdx.x; i < D; i += 256) { const float v = x[i]; ss += v * v; }
  red[threadIdx.x] = ss;
  __syncthreads();
  for (int s = 128; s > 0; s >>= 1) {
    if (threadIdx.x < s) red[threadIdx.x] += red[threadIdx.x + s];
    __syncthreads();
  }
  const float inv = rsqrtf(red[0] / (float)D + 1e-6f);
  for (int i = threadIdx.x; i < D; i += 256)
    dstB[(size_t)row * bld + i] = f2bf(x[i] * inv * w[i]);
}

__global__ __launch_bounds__(256) void rope_kernel(
    const int* __restrict__ pos,
    const float* __restrict__ src, int srow, int shead,
    unsigned short* __restrict__ dst, int drow, int dhead,
    int H, int T)
{
  const int idx = blockIdx.x * 256 + threadIdx.x;
  const int total = T * H * 32;
  if (idx >= total) return;
  const int i = idx & 31;
  const int h = (idx >> 5) % H;
  const int t = idx / (32 * H);
  const float invf = __expf(-9.210340371976184f * ((float)(2 * i) / 64.0f));
  const float ang = (float)pos[t] * invf;
  const float c = __cosf(ang), s = __sinf(ang);
  const float* p = src + (size_t)t * srow + (size_t)h * shead;
  const float x1 = p[2 * i], x2 = p[2 * i + 1];
  unsigned short* q = dst + (size_t)t * drow + (size_t)h * dhead;
  q[2 * i]     = f2bf(x1 * c - x2 * s);
  q[2 * i + 1] = f2bf(x2 * c + x1 * s);
}

__global__ __launch_bounds__(256) void cvt_bf16_kernel(
    const float* __restrict__ src, int sld,
    unsigned short* __restrict__ dst, int dld, int rows, int cols)
{
  const int idx = blockIdx.x * 256 + threadIdx.x;
  if (idx >= rows * cols) return;
  const int r = idx / cols, c = idx - r * cols;
  dst[(size_t)r * dld + c] = f2bf(src[(size_t)r * sld + c]);
}

__global__ __launch_bounds__(256) void cvt_t_bf16_kernel(
    const float* __restrict__ src, int sld,
    unsigned short* __restrict__ dst, int dld, int rows, int cols)
{
  const int idx = blockIdx.x * 256 + threadIdx.x;
  if (idx >= rows * cols) return;
  const int r = idx / cols, c = idx - r * cols;
  dst[(size_t)c * dld + r] = f2bf(src[(size_t)r * sld + c]);
}

__global__ __launch_bounds__(256) void tr_bf16_kernel(
    const unsigned short* __restrict__ src, int sld,
    unsigned short* __restrict__ dst, int dld, int rows, int cols)
{
  const int idx = blockIdx.x * 256 + threadIdx.x;
  if (idx >= rows * cols) return;
  const int r = idx / cols, c = idx - r * cols;
  dst[(size_t)c * dld + r] = src[(size_t)r * sld + c];
}

__global__ __launch_bounds__(256) void add_kernel(
    const float* __restrict__ a, const float* __restrict__ b,
    float* __restrict__ c, int n)
{
  const int idx = blockIdx.x * 256 + threadIdx.x;
  if (idx < n) c[idx] = a[idx] + b[idx];
}

__global__ __launch_bounds__(256) void silu_mul_kernel(
    unsigned short* __restrict__ g, const unsigned short* __restrict__ u, int n)
{
  const int idx = blockIdx.x * 256 + threadIdx.x;
  if (idx >= n) return;
  const float gv = bf2f(g[idx]);
  const float uv = bf2f(u[idx]);
  const float s = gv / (1.0f + __expf(-gv));
  g[idx] = f2bf(s * uv);
}

// ---------------------------------------------------------------------------
// Host orchestration
// ---------------------------------------------------------------------------
extern "C" void kernel_launch(void* const* d_in, const int* in_sizes, int n_in,
                              void* d_out, int out_size, void* d_ws, size_t ws_size,
                              hipStream_t stream) {
  (void)in_sizes; (void)n_in; (void)out_size; (void)ws_size;

  const int T = 2048, HID = 2048, QLR = 1536, KVLR = 512, NH = 16;
  const int DN = 128, DR = 64, DV = 128, INTER = 10944;
  const int DQ = DN + DR;                 // 192
  const int DKF = KVLR + DR;              // 576
  const float SCALE = 0.07216878364870323f;   // (DN+DR)^-0.5

  const int* positions   = (const int*)  d_in[0];
  const float* hidden    = (const float*)d_in[1];
  const float* input_ln  = (const float*)d_in[2];
  const float* q_a_w     = (const float*)d_in[3];
  const float* q_a_ln    = (const float*)d_in[4];
  const float* q_b_w     = (const float*)d_in[5];
  const float* kv_a_w    = (const float*)d_in[6];
  const float* kv_a_ln   = (const float*)d_in[7];
  const float* w_uk      = (const float*)d_in[8];
  const float* w_uv      = (const float*)d_in[9];
  const float* o_w       = (const float*)d_in[10];
  const float* post_ln   = (const float*)d_in[11];
  const float* gate_w    = (const float*)d_in[12];
  const float* up_w      = (const float*)d_in[13];
  const float* down_w    = (const float*)d_in[14];

  char* ws = (char*)d_ws;
  size_t off = 0;
  auto alloc = [&](size_t bytes) -> char* {
    char* p = ws + off;
    off += (bytes + 255) & ~(size_t)255;
    return p;
  };

  // persistent across both phases
  float* h2 = (float*)alloc((size_t)T * HID * 4);
  const size_t arena = off;

  // ---------------- attention phase scratch ----------------
  unsigned short* h_bf   = (unsigned short*)alloc((size_t)T * HID * 2);
  float*          qa_f   = (float*)         alloc((size_t)T * QLR * 4);
  unsigned short* qawt   = (unsigned short*)alloc((size_t)QLR * HID * 2);
  unsigned short* qc_bf  = (unsigned short*)alloc((size_t)T * QLR * 2);
  unsigned short* qbwt   = (unsigned short*)alloc((size_t)(NH * DQ) * QLR * 2);
  float*          q_f    = (float*)         alloc((size_t)T * NH * DQ * 4);
  unsigned short* q_bf   = (unsigned short*)alloc((size_t)T * NH * DQ * 2);
  unsigned short* kvawt  = (unsigned short*)alloc((size_t)DKF * HID * 2);
  float*          kva_f  = (float*)         alloc((size_t)T * DKF * 4);
  unsigned short* kfull  = (unsigned short*)alloc((size_t)T * DKF * 2);
  unsigned short* compt  = (unsigned short*)alloc((size_t)KVLR * T * 2);
  unsigned short* wuk_bf = (unsigned short*)alloc((size_t)KVLR * NH * DN * 2);
  unsigned short* qfull  = (unsigned short*)alloc((size_t)T * NH * DKF * 2);
  unsigned short* ol_bf  = (unsigned short*)alloc((size_t)T * NH * KVLR * 2);
  unsigned short* wuvt   = (unsigned short*)alloc((size_t)NH * DV * KVLR * 2);
  unsigned short* ov_bf  = (unsigned short*)alloc((size_t)T * NH * DV * 2);
  unsigned short* owt    = (unsigned short*)alloc((size_t)HID * (NH * DV) * 2);
  float*          aout_f = (float*)         alloc((size_t)T * HID * 4);

  auto cdiv = [](long long a, long long b) { return (int)((a + b - 1) / b); };
  dim3 blk(256);

  auto gemm = [&](const unsigned short* A, int lda, const unsigned short* Bt,
                  int ldb, void* C, int ldc, int M, int N, int K, int bf) {
    dim3 grid(cdiv(N, 128), M / 128);
    gemm_bf16_kernel<<<grid, blk, 0, stream>>>(A, lda, Bt, ldb, C, ldc, M, N, K, bf);
  };

  // h = rmsnorm(hidden, input_ln)  -> bf16
  rmsnorm_kernel<<<T, blk, 0, stream>>>(hidden, HID, input_ln, h_bf, HID, HID);

  // q_a = h @ q_a_w
  cvt_t_bf16_kernel<<<cdiv((long long)HID * QLR, 256), blk, 0, stream>>>(
      q_a_w, QLR, qawt, HID, HID, QLR);
  gemm(h_bf, HID, qawt, HID, qa_f, QLR, T, QLR, HID, 0);

  // qc = rmsnorm(q_a, q_a_ln) -> bf16
  rmsnorm_kernel<<<T, blk, 0, stream>>>(qa_f, QLR, q_a_ln, qc_bf, QLR, QLR);

  // q = qc @ q_b_w   (f32, then bf16 copy for the absorb GEMM)
  cvt_t_bf16_kernel<<<cdiv((long long)QLR * NH * DQ, 256), blk, 0, stream>>>(
      q_b_w, NH * DQ, qbwt, QLR, QLR, NH * DQ);
  gemm(qc_bf, QLR, qbwt, QLR, q_f, NH * DQ, T, NH * DQ, QLR, 0);
  cvt_bf16_kernel<<<cdiv((long long)T * NH * DQ, 256), blk, 0, stream>>>(
      q_f, NH * DQ, q_bf, NH * DQ, T, NH * DQ);

  // kv_a = h @ kv_a_w
  cvt_t_bf16_kernel<<<cdiv((long long)HID * DKF, 256), blk, 0, stream>>>(
      kv_a_w, DKF, kvawt, HID, HID, DKF);
  gemm(h_bf, HID, kvawt, HID, kva_f, DKF, T, DKF, HID, 0);

  // compressed = rmsnorm(kv_a[:, :512]) -> k_full[:, :512] bf16
  rmsnorm_kernel<<<T, blk, 0, stream>>>(kva_f, DKF, kv_a_ln, kfull, DKF, KVLR);
  // compressed^T (512 x T) for the P x C GEMM B-operand
  tr_bf16_kernel<<<cdiv((long long)T * KVLR, 256), blk, 0, stream>>>(
      kfull, DKF, compt, T, T, KVLR);

  // RoPE: q_rope -> q_full[:, :, 512:576]; k_rope -> k_full[:, 512:576]
  rope_kernel<<<cdiv((long long)T * NH * 32, 256), blk, 0, stream>>>(
      positions, q_f + DN, NH * DQ, DQ, qfull + KVLR, NH * DKF, DKF, NH, T);
  rope_kernel<<<cdiv((long long)T * 32, 256), blk, 0, stream>>>(
      positions, kva_f + KVLR, DKF, 0, kfull + KVLR, DKF, 0, 1, T);

  // ql_nope[t,h,r] = sum_d q_nope[t,h,d] * w_uk[r,h,d]  -> q_full[:, :, :512]
  cvt_bf16_kernel<<<cdiv((long long)KVLR * NH * DN, 256), blk, 0, stream>>>(
      w_uk, NH * DN, wuk_bf, NH * DN, KVLR, NH * DN);
  for (int h = 0; h < NH; ++h) {
    gemm(q_bf + h * DQ, NH * DQ, wuk_bf + h * DN, NH * DN,
         qfull + h * DKF, NH * DKF, T, KVLR, DN, 1);
  }

  // attention: o_latent bf16 (T, NH, 512)
  {
    dim3 grid(T / AQ, NH);
    mla_attn_kernel<<<grid, blk, 0, stream>>>(qfull, kfull, compt, ol_bf, T, SCALE);
  }

  // o_v[t,h,d] = sum_r o_latent[t,h,r] * w_uv[r,h,d]
  for (int h = 0; h < NH; ++h) {
    cvt_t_bf16_kernel<<<cdiv((long long)KVLR * DV, 256), blk, 0, stream>>>(
        w_uv + h * DV, NH * DV, wuvt + (size_t)h * DV * KVLR, KVLR, KVLR, DV);
  }
  for (int h = 0; h < NH; ++h) {
    gemm(ol_bf + h * KVLR, NH * KVLR, wuvt + (size_t)h * DV * KVLR, KVLR,
         ov_bf + h * DV, NH * DV, T, DV, KVLR, 1);
  }

  // attn_out = o_v @ o_w ; h2 = hidden + attn_out
  cvt_t_bf16_kernel<<<cdiv((long long)NH * DV * HID, 256), blk, 0, stream>>>(
      o_w, HID, owt, NH * DV, NH * DV, HID);
  gemm(ov_bf, NH * DV, owt, NH * DV, aout_f, HID, T, HID, NH * DV, 0);
  add_kernel<<<cdiv((long long)T * HID, 256), blk, 0, stream>>>(
      hidden, aout_f, h2, T * HID);

  // ---------------- MLP phase (reuse arena) ----------------
  off = arena;
  unsigned short* hn_bf   = (unsigned short*)alloc((size_t)T * HID * 2);
  unsigned short* gatewt  = (unsigned short*)alloc((size_t)INTER * HID * 2);
  unsigned short* upwt    = (unsigned short*)alloc((size_t)INTER * HID * 2);
  unsigned short* downwt  = (unsigned short*)alloc((size_t)HID * INTER * 2);
  unsigned short* gate_bf = (unsigned short*)alloc((size_t)T * INTER * 2);
  unsigned short* up_bf   = (unsigned short*)alloc((size_t)T * INTER * 2);
  float*          mlp_f   = (float*)         alloc((size_t)T * HID * 4);

  rmsnorm_kernel<<<T, blk, 0, stream>>>(h2, HID, post_ln, hn_bf, HID, HID);

  cvt_t_bf16_kernel<<<cdiv((long long)HID * INTER, 256), blk, 0, stream>>>(
      gate_w, INTER, gatewt, HID, HID, INTER);
  gemm(hn_bf, HID, gatewt, HID, gate_bf, INTER, T, INTER, HID, 1);

  cvt_t_bf16_kernel<<<cdiv((long long)HID * INTER, 256), blk, 0, stream>>>(
      up_w, INTER, upwt, HID, HID, INTER);
  gemm(hn_bf, HID, upwt, HID, up_bf, INTER, T, INTER, HID, 1);

  silu_mul_kernel<<<cdiv((long long)T * INTER, 256), blk, 0, stream>>>(
      gate_bf, up_bf, T * INTER);

  cvt_t_bf16_kernel<<<cdiv((long long)INTER * HID, 256), blk, 0, stream>>>(
      down_w, HID, downwt, INTER, INTER, HID);
  gemm(gate_bf, INTER, downwt, INTER, mlp_f, HID, T, HID, INTER, 0);

  add_kernel<<<cdiv((long long)T * HID, 256), blk, 0, stream>>>(
      h2, mlp_f, (float*)d_out, T * HID);
}
